// EHGN_TopK_Dense_51453708206090
// MI455X (gfx1250) — matrix operand order repaired and checked by reading, not compiled, not run on hardware
//
#include <hip/hip_runtime.h>
#include <hip/hip_bf16.h>

// ---------------------------------------------------------------------------
// EHGN (EGNN + differentiable pooling) for MI455X / gfx1250.
// All dense MLP layers run on v_wmma_f32_16x16x32_bf16 (bf16 in, f32 acc).
// Weights are pre-swizzled into WMMA-fragment-native layout so each lane's
// B-fragment is ONE 32-byte contiguous load (2x global_load_b128) instead of
// 16 strided global_load_u16s. Activation staging uses float4 loads and
// 8-byte LDS stores. The per-edge pipeline (gather -> phi_e -> agg scatter ->
// phi_x -> coord scatter) is fused per 16-edge tile so the [E x 128] message
// tensor stays in LDS and never hits HBM.
// ---------------------------------------------------------------------------

typedef __attribute__((ext_vector_type(16))) __bf16 v16bf;
typedef __attribute__((ext_vector_type(4)))  __bf16 v4bf;
typedef __attribute__((ext_vector_type(8)))  float  v8f;

__device__ __forceinline__ float silu_f(float v) {
  return v / (1.0f + __expf(-v));
}

// ---- WMMA fragment helpers (CDNA5 16-bit layouts, CDNA5 ISA) ----
// A 16x32 from row-major LDS: lanes 0-15 row M=lane, K-halves 0/16; lanes
// 16-31 same rows, K halves 8/24. Two contiguous 8-element runs per lane
// -> 2x ds_load_b128.
__device__ __forceinline__ v16bf frag_a(const __bf16* a0, int lda, int lane) {
  const int r  = lane & 15;
  const int kb = (lane & 16) ? 8 : 0;
  const __bf16* p = a0 + (size_t)r * lda + kb;
  v16bf f;
#pragma unroll
  for (int v = 0; v < 8; ++v) f[v]     = p[v];
#pragma unroll
  for (int v = 0; v < 8; ++v) f[8 + v] = p[16 + v];
  return f;
}

// One 16x16 output tile of A[16xK] @ W[KxN].
// W is pre-swizzled: per 32x16 (KxN) tile, 32 lanes x 16 contiguous values.
__device__ __forceinline__ v8f gemm_tile(const __bf16* A, int lda,
                                         const __bf16* Wsw, int Nn,
                                         int K, int nc, int lane, v8f acc) {
  const size_t kstride = (size_t)(Nn >> 4) * 512;  // elements per K-tile row
  const __bf16* wp = Wsw + ((size_t)(nc >> 4) * 32 + lane) * 16;
  for (int kc = 0; kc < K; kc += 32) {
    if (kc + 32 < K)  // global_prefetch_b8 of the next weight slab
      __builtin_prefetch(wp + kstride, 0, 1);
    v16bf fa = frag_a(A + kc, lda, lane);
    v16bf fb = *(const v16bf*)wp;                  // 32B contiguous per lane
    acc = __builtin_amdgcn_wmma_f32_16x16x32_bf16(
        false, fa, false, fb, (short)0, acc, false, false);
    wp += kstride;
  }
  return acc;
}

// C/D tile: lane = column N, VGPR g = row (g or g+8 by lane half).
__device__ __forceinline__ void tile_store_silu(v8f acc, const float* bias,
                                                int nc, __bf16* out, int ldo,
                                                int lane) {
  const int n  = lane & 15;
  const int mb = (lane & 16) ? 8 : 0;
  const float bv = bias[nc + n];
#pragma unroll
  for (int g = 0; g < 8; ++g)
    out[(size_t)(mb + g) * ldo + nc + n] = (__bf16)silu_f(acc[g] + bv);
}

// ---- weight convert: f32 row-major [K x N] -> bf16 fragment-swizzled -------
// dst[((kt*NT + nt)*32 + lane)*16 + j] = W[kt*32 + kb(lane) + (j&7) + (j>=8?16:0)][nt*16 + c]
__global__ void k_cvt_w_frag(const float* __restrict__ src,
                             __bf16* __restrict__ dst, int Kdim, int Nn,
                             int Kp) {
  int idx = blockIdx.x * 256 + threadIdx.x;
  int tot = Kp * Nn;
  if (idx >= tot) return;
  const int NT = Nn >> 4;
  int t    = idx >> 9;        // 512 elements per 32x16 tile
  int r    = idx & 511;
  int lane = r >> 4;
  int j    = r & 15;
  int kt = t / NT, nt = t - kt * NT;
  int c  = lane & 15;
  int kb = (lane & 16) ? 8 : 0;
  int k    = kt * 32 + kb + (j & 7) + ((j & 8) << 1);
  int ncol = nt * 16 + c;
  float v = (k < Kdim) ? src[(size_t)k * Nn + ncol] : 0.0f;
  dst[idx] = (__bf16)v;
}

// ---------------------------------------------------------------------------
// Fused edge kernel: 16 edges/block, 256 threads (8 waves x 16 out columns).
// phi_e (273->128 SiLU ->128 SiLU) + agg scatter, phi_x (128->128 SiLU ->1)
// + equivariant coordinate scatter. All GEMMs via WMMA, messages in LDS.
// ---------------------------------------------------------------------------
__global__ __launch_bounds__(256)
void k_edge_layer(const int* __restrict__ row, const int* __restrict__ col,
                  const float* __restrict__ xin, const float* __restrict__ hin,
                  const float* __restrict__ eattr,
                  const __bf16* __restrict__ We0, const float* __restrict__ be0,
                  const __bf16* __restrict__ We1, const float* __restrict__ be1,
                  const __bf16* __restrict__ Wx0, const float* __restrict__ bx0,
                  const float* __restrict__ Wx1f, const float* __restrict__ bx1,
                  float* __restrict__ xout, float* __restrict__ agg, int E) {
  __shared__ __bf16 Af[16 * 288];   // [h_i | h_j | d2 | e_ij] padded 273->288
  __shared__ __bf16 M1[16 * 128];
  __shared__ __bf16 M2[16 * 128];   // message m (post-SiLU)
  __shared__ __bf16 P1[16 * 128];
  __shared__ float  relb[16 * 4];   // rel xyz + d2
  __shared__ int    er[16];
  __shared__ int    ec[16];

  const int tid   = threadIdx.x;
  const int lane  = tid & 31;
  const int wave  = tid >> 5;
  const int ebase = blockIdx.x * 16;

  if (tid < 16) {
    int e = ebase + tid;
    int r = 0, c = 0;
    if (e < E) { r = row[e]; c = col[e]; }
    er[tid] = r; ec[tid] = c;
    float d2 = 0.0f;
    for (int d = 0; d < 3; ++d) {
      float rv = xin[(size_t)r * 3 + d] - xin[(size_t)c * 3 + d];
      relb[tid * 4 + d] = rv;
      d2 += rv * rv;
    }
    relb[tid * 4 + 3] = d2;
  }
  __syncthreads();

  // Feature gather, float4-vectorized (72 chunks of 4 per edge row).
  for (int idx = tid; idx < 16 * 72; idx += 256) {
    int te = idx / 72;
    int k4 = (idx - te * 72) * 4;
    int e = ebase + te;
    float f0 = 0.f, f1 = 0.f, f2 = 0.f, f3 = 0.f;
    if (e < E) {
      if (k4 < 128) {
        const float4 v = *(const float4*)(hin + (size_t)er[te] * 128 + k4);
        f0 = v.x; f1 = v.y; f2 = v.z; f3 = v.w;
      } else if (k4 < 256) {
        const float4 v = *(const float4*)(hin + (size_t)ec[te] * 128 + (k4 - 128));
        f0 = v.x; f1 = v.y; f2 = v.z; f3 = v.w;
      } else {
        float t4[4];
#pragma unroll
        for (int j = 0; j < 4; ++j) {
          int k = k4 + j;
          t4[j] = (k == 256) ? relb[te * 4 + 3]
                : (k < 273)  ? eattr[(size_t)e * 16 + (k - 257)]
                             : 0.0f;
        }
        f0 = t4[0]; f1 = t4[1]; f2 = t4[2]; f3 = t4[3];
      }
    }
    v4bf o;
    o[0] = (__bf16)f0; o[1] = (__bf16)f1; o[2] = (__bf16)f2; o[3] = (__bf16)f3;
    *(v4bf*)(Af + te * 288 + k4) = o;
  }
  __syncthreads();

  const int nc = wave * 16;
  {   // phi_e layer 0
    v8f acc = {};
    acc = gemm_tile(Af, 288, We0, 128, 288, nc, lane, acc);
    tile_store_silu(acc, be0, nc, M1, 128, lane);
  }
  __syncthreads();
  {   // phi_e layer 1 (final_act=True) + agg[row] += m
    v8f acc = {};
    acc = gemm_tile(M1, 128, We1, 128, 128, nc, lane, acc);
    const int n  = lane & 15;
    const int mb = (lane & 16) ? 8 : 0;
    const float bv = be1[nc + n];
#pragma unroll
    for (int g = 0; g < 8; ++g) {
      int m = mb + g;
      float v = silu_f(acc[g] + bv);
      M2[m * 128 + nc + n] = (__bf16)v;
      int e = ebase + m;
      if (e < E) atomicAdd(&agg[(size_t)er[m] * 128 + nc + n], v);
    }
  }
  __syncthreads();
  {   // phi_x layer 0
    v8f acc = {};
    acc = gemm_tile(M2, 128, Wx0, 128, 128, nc, lane, acc);
    tile_store_silu(acc, bx0, nc, P1, 128, lane);
  }
  __syncthreads();
  if (tid < 16) {   // phi_x layer 1 (128 -> 1) + coordinate update
    int e = ebase + tid;
    float s = bx1[0];
    for (int k = 0; k < 128; ++k) s += (float)P1[tid * 128 + k] * Wx1f[k];
    if (e < E) {
      for (int d = 0; d < 3; ++d)
        atomicAdd(&xout[(size_t)er[tid] * 3 + d], relb[tid * 4 + d] * s);
    }
  }
}

// ---------------------------------------------------------------------------
// Node kernel: h_out = h_in + phi_h([h_in, agg]).
// ---------------------------------------------------------------------------
__global__ __launch_bounds__(256)
void k_node_layer(const float* __restrict__ hin, const float* __restrict__ agg,
                  const __bf16* __restrict__ Wh0, const float* __restrict__ bh0,
                  const __bf16* __restrict__ Wh1, const float* __restrict__ bh1,
                  float* __restrict__ hout, int Ncnt) {
  __shared__ __bf16 Af[16 * 256];
  __shared__ __bf16 Mb[16 * 128];
  const int tid  = threadIdx.x;
  const int lane = tid & 31;
  const int wave = tid >> 5;
  const int nb   = blockIdx.x * 16;

  for (int idx = tid; idx < 16 * 64; idx += 256) {   // 64 float4 chunks / row
    int tn = idx >> 6;
    int k4 = (idx & 63) * 4;
    int nd = nb + tn;
    float4 v = make_float4(0.f, 0.f, 0.f, 0.f);
    if (nd < Ncnt)
      v = (k4 < 128) ? *(const float4*)(hin + (size_t)nd * 128 + k4)
                     : *(const float4*)(agg + (size_t)nd * 128 + (k4 - 128));
    v4bf o;
    o[0] = (__bf16)v.x; o[1] = (__bf16)v.y; o[2] = (__bf16)v.z; o[3] = (__bf16)v.w;
    *(v4bf*)(Af + tn * 256 + k4) = o;
  }
  __syncthreads();

  const int nc = wave * 16;
  {
    v8f acc = {};
    acc = gemm_tile(Af, 256, Wh0, 128, 256, nc, lane, acc);
    tile_store_silu(acc, bh0, nc, Mb, 128, lane);
  }
  __syncthreads();
  {
    v8f acc = {};
    acc = gemm_tile(Mb, 128, Wh1, 128, 128, nc, lane, acc);
    const int n  = lane & 15;
    const int mb = (lane & 16) ? 8 : 0;
    const float bv = bh1[nc + n];
#pragma unroll
    for (int g = 0; g < 8; ++g) {
      int nd = nb + mb + g;
      if (nd < Ncnt)
        hout[(size_t)nd * 128 + nc + n] =
            hin[(size_t)nd * 128 + nc + n] + acc[g] + bv;
    }
  }
}

// ---------------------------------------------------------------------------
// Pooling MLP 128 -> 1024 -> 128 -> 5 + softmax; [16 x 1024] hidden in LDS.
// ---------------------------------------------------------------------------
__global__ __launch_bounds__(256)
void k_pool(const float* __restrict__ h,
            const __bf16* __restrict__ W0, const float* __restrict__ b0,
            const __bf16* __restrict__ W1, const float* __restrict__ b1,
            const float* __restrict__ W2f, const float* __restrict__ b2,
            float* __restrict__ pooling, int Ncnt) {
  __shared__ __bf16 A0[16 * 128];
  __shared__ __bf16 A1[16 * 1024];
  __shared__ __bf16 A2[16 * 128];
  __shared__ float  Lg[16 * 8];
  const int tid  = threadIdx.x;
  const int lane = tid & 31;
  const int wave = tid >> 5;
  const int nb   = blockIdx.x * 16;

  for (int idx = tid; idx < 16 * 32; idx += 256) {
    int tn = idx >> 5;
    int k4 = (idx & 31) * 4;
    int nd = nb + tn;
    float4 v = make_float4(0.f, 0.f, 0.f, 0.f);
    if (nd < Ncnt) v = *(const float4*)(h + (size_t)nd * 128 + k4);
    v4bf o;
    o[0] = (__bf16)v.x; o[1] = (__bf16)v.y; o[2] = (__bf16)v.z; o[3] = (__bf16)v.w;
    *(v4bf*)(A0 + tn * 128 + k4) = o;
  }
  __syncthreads();

  for (int t = 0; t < 8; ++t) {      // layer 0: 64 column tiles / 8 waves
    int nc = wave * 128 + t * 16;
    v8f acc = {};
    acc = gemm_tile(A0, 128, W0, 1024, 128, nc, lane, acc);
    tile_store_silu(acc, b0, nc, A1, 1024, lane);
  }
  __syncthreads();
  {                                   // layer 1: K = 1024
    int nc = wave * 16;
    v8f acc = {};
    acc = gemm_tile(A1, 1024, W1, 128, 1024, nc, lane, acc);
    tile_store_silu(acc, b1, nc, A2, 128, lane);
  }
  __syncthreads();
  if (tid < 80) {                     // layer 2: 128 -> 5 (scalar)
    int te = tid / 5, j = tid - te * 5;
    float s = b2[j];
    for (int k = 0; k < 128; ++k) s += (float)A2[te * 128 + k] * W2f[k * 5 + j];
    Lg[te * 8 + j] = s;
  }
  __syncthreads();
  if (tid < 16) {                     // softmax over K=5
    int nd = nb + tid;
    if (nd < Ncnt) {
      float mx = Lg[tid * 8 + 0];
      for (int j = 1; j < 5; ++j) mx = fmaxf(mx, Lg[tid * 8 + j]);
      float sum = 0.0f, e5[5];
      for (int j = 0; j < 5; ++j) { e5[j] = __expf(Lg[tid * 8 + j] - mx); sum += e5[j]; }
      float inv = 1.0f / sum;
      for (int j = 0; j < 5; ++j) pooling[(size_t)nd * 5 + j] = e5[j] * inv;
    }
  }
}

// ---- cluster pooling: count, X = s_t@x/cnt, H = s_t@h_l/cnt (per b,k) ------
__global__ __launch_bounds__(128)
void k_cluster(const float* __restrict__ pooling, const float* __restrict__ x1,
               const float* __restrict__ h_l, float* __restrict__ X,
               float* __restrict__ H, int n, int K) {
  const int bk = blockIdx.x;
  const int b = bk / K, k = bk - b * K;
  __shared__ float red[128];
  const int t = threadIdx.x;
  float c = 0.0f;
  for (int i = t; i < n; i += 128) c += pooling[((size_t)b * n + i) * K + k];
  red[t] = c;
  __syncthreads();
  for (int s = 64; s > 0; s >>= 1) {
    if (t < s) red[t] += red[t + s];
    __syncthreads();
  }
  const float cnt = fmaxf(red[0], 1e-5f);
  float s = 0.0f;
  for (int i = 0; i < n; ++i)
    s += pooling[((size_t)b * n + i) * K + k] * h_l[((size_t)b * n + i) * 128 + t];
  H[(size_t)bk * 128 + t] = s / cnt;
  if (t < 3) {
    float sx = 0.0f;
    for (int i = 0; i < n; ++i)
      sx += pooling[((size_t)b * n + i) * K + k] * x1[((size_t)b * n + i) * 3 + t];
    X[(size_t)bk * 3 + t] = sx / cnt;
  }
}

// ---- a_node[row] += pooling[col] (sparse adjacency @ pooling) --------------
__global__ void k_anode(const int* __restrict__ row, const int* __restrict__ col,
                        const float* __restrict__ pooling,
                        float* __restrict__ anode, int E) {
  int e = blockIdx.x * 256 + threadIdx.x;
  if (e >= E) return;
  int r = row[e], c = col[e];
  for (int j = 0; j < 5; ++j)
    atomicAdd(&anode[(size_t)r * 5 + j], pooling[(size_t)c * 5 + j]);
}

// ---- a[b,i,j] = sum_n s_t[b,i,n] * a_node[b,n,j] ---------------------------
__global__ void k_adj(const float* __restrict__ pooling,
                      const float* __restrict__ anode, float* __restrict__ a,
                      int n, int K) {
  int b = blockIdx.x, t = threadIdx.x;
  if (t >= K * K) return;
  int i = t / K, j = t - i * K;
  float s = 0.0f;
  for (int nn = 0; nn < n; ++nn) {
    size_t nd = (size_t)b * n + nn;
    s += pooling[nd * K + i] * anode[nd * K + j];
  }
  a[(size_t)b * K * K + t] = s;
}

// ---- coarse edge construction + mlp_1 (1 -> 128 -> 128 -> 16) --------------
__global__ __launch_bounds__(128)
void k_coarse_edges(const float* __restrict__ a,
                    const float* __restrict__ W0, const float* __restrict__ b0,
                    const float* __restrict__ W1, const float* __restrict__ b1,
                    const float* __restrict__ W2, const float* __restrict__ b2,
                    int* __restrict__ rowC, int* __restrict__ colC,
                    float* __restrict__ eaC, int K) {
  const int e = blockIdx.x;
  const int per = K * (K - 1);
  const int b = e / per, p = e - b * per;
  const int i = p / (K - 1), jo = p - i * (K - 1);
  const int j = jo + (jo >= i ? 1 : 0);
  const float attr = a[(size_t)b * K * K + i * K + j];
  if (threadIdx.x == 0) { rowC[e] = b * K + i; colC[e] = b * K + j; }
  __shared__ float t1[128];
  __shared__ float t2[128];
  const int t = threadIdx.x;
  t1[t] = silu_f(attr * W0[t] + b0[t]);
  __syncthreads();
  float s = b1[t];
  for (int k = 0; k < 128; ++k) s += t1[k] * W1[k * 128 + t];
  t2[t] = silu_f(s);
  __syncthreads();
  if (t < 16) {
    float s2 = b2[t];
    for (int k = 0; k < 128; ++k) s2 += t2[k] * W2[k * 16 + t];
    eaC[(size_t)e * 16 + t] = s2;
  }
}

// ---- upsample: hO = h_l + 0.5 * (s @ H'), xO = (x_l + Xx)/2, Xx = s @ X' ---
__global__ void k_upsample_h(const float* __restrict__ h_l,
                             const float* __restrict__ pooling,
                             const float* __restrict__ H,
                             float* __restrict__ hO, int Ncnt, int n, int K) {
  int idx = blockIdx.x * 256 + threadIdx.x;
  if (idx >= Ncnt * 128) return;
  int nd = idx >> 7, d = idx & 127;
  int b = nd / n;
  float s = 0.0f;
  for (int k = 0; k < 5; ++k)
    s += pooling[(size_t)nd * 5 + k] * H[((size_t)(b * K + k)) * 128 + d];
  hO[idx] = h_l[idx] + 0.5f * s;
}

__global__ void k_upsample_x(const float* __restrict__ x_l,
                             const float* __restrict__ pooling,
                             const float* __restrict__ X,
                             float* __restrict__ Xx, float* __restrict__ xO,
                             int Ncnt, int n, int K) {
  int idx = blockIdx.x * 256 + threadIdx.x;
  if (idx >= Ncnt * 3) return;
  int nd = idx / 3, d = idx - nd * 3;
  int b = nd / n;
  float s = 0.0f;
  for (int k = 0; k < 5; ++k)
    s += pooling[(size_t)nd * 5 + k] * X[(size_t)(b * K + k) * 3 + d];
  Xx[idx] = s;
  xO[idx] = 0.5f * (x_l[idx] + s);
}

// ---- final: result_x = (x_l - Xx)/2 * mlp_2(rh) + Xx (mlp_2 via WMMA) ------
__global__ __launch_bounds__(256)
void k_final(const float* __restrict__ rh, const float* __restrict__ x_l,
             const float* __restrict__ Xx,
             const __bf16* __restrict__ W0, const float* __restrict__ b0,
             const __bf16* __restrict__ W1, const float* __restrict__ b1,
             const float* __restrict__ W2f, const float* __restrict__ b2,
             float* __restrict__ out_x, int Ncnt) {
  __shared__ __bf16 A0[16 * 128];
  __shared__ __bf16 A1[16 * 128];
  __shared__ __bf16 A2[16 * 128];
  const int tid  = threadIdx.x;
  const int lane = tid & 31;
  const int wave = tid >> 5;
  const int nb   = blockIdx.x * 16;

  for (int idx = tid; idx < 16 * 32; idx += 256) {
    int tn = idx >> 5;
    int k4 = (idx & 31) * 4;
    int nd = nb + tn;
    float4 v = make_float4(0.f, 0.f, 0.f, 0.f);
    if (nd < Ncnt) v = *(const float4*)(rh + (size_t)nd * 128 + k4);
    v4bf o;
    o[0] = (__bf16)v.x; o[1] = (__bf16)v.y; o[2] = (__bf16)v.z; o[3] = (__bf16)v.w;
    *(v4bf*)(A0 + tn * 128 + k4) = o;
  }
  __syncthreads();
  const int nc = wave * 16;
  {
    v8f acc = {};
    acc = gemm_tile(A0, 128, W0, 128, 128, nc, lane, acc);
    tile_store_silu(acc, b0, nc, A1, 128, lane);
  }
  __syncthreads();
  {
    v8f acc = {};
    acc = gemm_tile(A1, 128, W1, 128, 128, nc, lane, acc);
    tile_store_silu(acc, b1, nc, A2, 128, lane);
  }
  __syncthreads();
  if (tid < 48) {
    int te = tid / 3, d = tid - te * 3;
    float g = b2[d];
    for (int k = 0; k < 128; ++k) g += (float)A2[te * 128 + k] * W2f[k * 3 + d];
    int nd = nb + te;
    if (nd < Ncnt) {
      size_t o = (size_t)nd * 3 + d;
      out_x[o] = 0.5f * (x_l[o] - Xx[o]) * g + Xx[o];
    }
  }
}

// ---------------------------------------------------------------------------
// Host orchestration
// ---------------------------------------------------------------------------
struct EdgeLayerW {
  const __bf16 *We0, *We1, *Wx0, *Wh0, *Wh1;
  const float *be0, *be1, *bx0, *Wx1f, *bx1, *bh0, *bh1;
};
struct EgnnW { EdgeLayerW L[2]; };

static void run_egnn(hipStream_t s, const EgnnW& W, const int* row,
                     const int* col, const float* eattr, float* xA, float* xB,
                     float* hA, float* hB, float* agg, int E, int Ncnt) {
  float *xi = xA, *xo = xB, *hi = hA, *ho = hB;
  for (int l = 0; l < 2; ++l) {
    const EdgeLayerW& w = W.L[l];
    hipMemcpyAsync(xo, xi, (size_t)Ncnt * 3 * sizeof(float),
                   hipMemcpyDeviceToDevice, s);
    hipMemsetAsync(agg, 0, (size_t)Ncnt * 128 * sizeof(float), s);
    k_edge_layer<<<(E + 15) / 16, 256, 0, s>>>(
        row, col, xi, hi, eattr, w.We0, w.be0, w.We1, w.be1, w.Wx0, w.bx0,
        w.Wx1f, w.bx1, xo, agg, E);
    k_node_layer<<<(Ncnt + 15) / 16, 256, 0, s>>>(hi, agg, w.Wh0, w.bh0, w.Wh1,
                                                  w.bh1, ho, Ncnt);
    float* t = xi; xi = xo; xo = t;
    t = hi; hi = ho; ho = t;
  }
  // L == 2 -> final state is back in xA / hA.
}

extern "C" void kernel_launch(void* const* d_in, const int* in_sizes, int n_in,
                              void* d_out, int out_size, void* d_ws,
                              size_t ws_size, hipStream_t stream) {
  (void)n_in; (void)out_size; (void)ws_size;
  const int*   edge_index = (const int*)d_in[0];
  const float* edge_attr  = (const float*)d_in[1];
  const float* x_l        = (const float*)d_in[2];
  const float* h_l        = (const float*)d_in[3];
  const int E = in_sizes[0] / 2;
  const int N = in_sizes[2] / 3;
  const int B = 64, K = 5;
  const int n = N / B;
  const int* row = edge_index;
  const int* col = edge_index + E;

  // Workspace bump allocator (deterministic layout each call).
  char* wp = (char*)d_ws;
  auto alloc = [&](size_t bytes) -> void* {
    void* r = (void*)wp;
    wp += (bytes + 255) & ~(size_t)255;
    return r;
  };
  auto F = [&](int i) { return (const float*)d_in[i]; };
  auto cvt = [&](const float* W, int Kdim, int Nn) -> const __bf16* {
    int Kp = (Kdim + 31) & ~31;
    __bf16* dst = (__bf16*)alloc((size_t)Kp * Nn * sizeof(__bf16));
    int tot = Kp * Nn;
    k_cvt_w_frag<<<(tot + 255) / 256, 256, 0, stream>>>(W, dst, Kdim, Nn, Kp);
    return dst;
  };

  // jax.tree flatten order (dict keys sorted): egnn_hidden(6..29),
  // egnn_in(30..53), egnn_out(54..77), mlp_1(78..83), mlp_2(84..89),
  // pool(90..95). Per egnn layer: phi_e{W0,b0,W1,b1}, phi_h{...}, phi_x{...}.
  auto build_egnn = [&](int base) {
    EgnnW w;
    for (int l = 0; l < 2; ++l) {
      int lb = base + l * 12;
      EdgeLayerW& e = w.L[l];
      e.We0 = cvt(F(lb + 0), 273, 128); e.be0 = F(lb + 1);
      e.We1 = cvt(F(lb + 2), 128, 128); e.be1 = F(lb + 3);
      e.Wh0 = cvt(F(lb + 4), 256, 128); e.bh0 = F(lb + 5);
      e.Wh1 = cvt(F(lb + 6), 128, 128); e.bh1 = F(lb + 7);
      e.Wx0 = cvt(F(lb + 8), 128, 128); e.bx0 = F(lb + 9);
      e.Wx1f = F(lb + 10);              e.bx1 = F(lb + 11);
    }
    return w;
  };
  EgnnW Whid = build_egnn(6);
  EgnnW Win  = build_egnn(30);
  EgnnW Wout = build_egnn(54);
  const float *Wm1_0 = F(78), *bm1_0 = F(79), *Wm1_1 = F(80), *bm1_1 = F(81),
              *Wm1_2 = F(82), *bm1_2 = F(83);
  const __bf16* Wm2_0 = cvt(F(84), 128, 128); const float* bm2_0 = F(85);
  const __bf16* Wm2_1 = cvt(F(86), 128, 128); const float* bm2_1 = F(87);
  const float *Wm2_2 = F(88), *bm2_2 = F(89);
  const __bf16* Wp0 = cvt(F(90), 128, 1024);  const float* bp0 = F(91);
  const __bf16* Wp1 = cvt(F(92), 1024, 128);  const float* bp1 = F(93);
  const float *Wp2 = F(94), *bp2 = F(95);

  // Activation / intermediate buffers.
  float* xA  = (float*)alloc((size_t)N * 3 * 4);
  float* xB  = (float*)alloc((size_t)N * 3 * 4);
  float* hA  = (float*)alloc((size_t)N * 128 * 4);
  float* hB  = (float*)alloc((size_t)N * 128 * 4);
  float* agg = (float*)alloc((size_t)N * 128 * 4);
  float* pooling = (float*)alloc((size_t)N * 5 * 4);
  float* anode   = (float*)alloc((size_t)N * 5 * 4);
  float* aC  = (float*)alloc((size_t)B * K * K * 4);
  float* Xc  = (float*)alloc((size_t)B * K * 3 * 4);
  float* xcB = (float*)alloc((size_t)B * K * 3 * 4);
  float* Hc  = (float*)alloc((size_t)B * K * 128 * 4);
  float* hcB = (float*)alloc((size_t)B * K * 128 * 4);
  float* aggC = (float*)alloc((size_t)B * K * 128 * 4);
  int*   rowC = (int*)alloc((size_t)B * K * (K - 1) * 4);
  int*   colC = (int*)alloc((size_t)B * K * (K - 1) * 4);
  float* eaC  = (float*)alloc((size_t)B * K * (K - 1) * 16 * 4);
  float* Xx  = (float*)alloc((size_t)N * 3 * 4);
  float* xoA = (float*)alloc((size_t)N * 3 * 4);
  float* xoB = (float*)alloc((size_t)N * 3 * 4);
  float* hoA = (float*)alloc((size_t)N * 128 * 4);
  float* hoB = (float*)alloc((size_t)N * 128 * 4);

  // 1) egnn_in on the full graph.
  hipMemcpyAsync(xA, x_l, (size_t)N * 3 * 4, hipMemcpyDeviceToDevice, stream);
  hipMemcpyAsync(hA, h_l, (size_t)N * 128 * 4, hipMemcpyDeviceToDevice, stream);
  run_egnn(stream, Win, row, col, edge_attr, xA, xB, hA, hB, agg, E, N);

  // 2) pooling assignments + cluster features.
  k_pool<<<(N + 15) / 16, 256, 0, stream>>>(hA, Wp0, bp0, Wp1, bp1, Wp2, bp2,
                                            pooling, N);
  k_cluster<<<B * K, 128, 0, stream>>>(pooling, xA, h_l, Xc, Hc, n, K);

  // 3) coarse adjacency a = s_t @ (A @ pooling) and coarse edge attrs.
  hipMemsetAsync(anode, 0, (size_t)N * 5 * 4, stream);
  k_anode<<<(E + 255) / 256, 256, 0, stream>>>(row, col, pooling, anode, E);
  k_adj<<<B, 32, 0, stream>>>(pooling, anode, aC, n, K);
  const int Ec = B * K * (K - 1);
  k_coarse_edges<<<Ec, 128, 0, stream>>>(aC, Wm1_0, bm1_0, Wm1_1, bm1_1, Wm1_2,
                                         bm1_2, rowC, colC, eaC, K);

  // 4) egnn_hidden on the coarse graph (same WMMA kernels, tiny grid).
  run_egnn(stream, Whid, rowC, colC, eaC, Xc, xcB, Hc, hcB, aggC, Ec, B * K);

  // 5) upsample and egnn_out on the full graph.
  k_upsample_x<<<(N * 3 + 255) / 256, 256, 0, stream>>>(x_l, pooling, Xc, Xx,
                                                        xoA, N, n, K);
  k_upsample_h<<<(N * 128 + 255) / 256, 256, 0, stream>>>(h_l, pooling, Hc,
                                                          hoA, N, n, K);
  run_egnn(stream, Wout, row, col, edge_attr, xoA, xoB, hoA, hoB, agg, E, N);

  // 6) final gating: result_x then rh, concatenated into d_out.
  float* out = (float*)d_out;
  k_final<<<(N + 15) / 16, 256, 0, stream>>>(hoA, x_l, Xx, Wm2_0, bm2_0, Wm2_1,
                                             bm2_1, Wm2_2, bm2_2, out, N);
  hipMemcpyAsync(out + (size_t)N * 3, hoA, (size_t)N * 128 * 4,
                 hipMemcpyDeviceToDevice, stream);
}
